// PredRNNv2_90580860272819
// MI455X (gfx1250) — compile-verified, matrix operand order checked by reading, not run on hardware
//
#include <hip/hip_runtime.h>
#include <hip/hip_bf16.h>
#include <math.h>

// ---------------------------------------------------------------------------
// PredRNN-v2 (3-layer ConvLSTM, NH=64, 48x48, B=4, T=20, horizon 4) on gfx1250
// Convs = implicit GEMM via v_wmma_f32_16x16x32_f16 (f16 operands, f32 acc).
// Weights pre-converted to f16 with K padded to 32 => unconditional b128 loads.
// ---------------------------------------------------------------------------

typedef __attribute__((ext_vector_type(16))) _Float16 v16h;
typedef __attribute__((ext_vector_type(8)))  _Float16 v8h;
typedef __attribute__((ext_vector_type(8)))  float    v8f;

#define BB       4
#define TT       20
#define TE       16          // encoder steps
#define HORIZON  4
#define HH       48
#define WW       48
#define HW       2304        // 48*48
#define NHID     64
#define NTOT     (BB*HW)     // 9216 output positions
#define NTILES   (NTOT/64)   // 144 N-tiles of 64

__device__ __forceinline__ float sigf(float x) { return 1.0f / (1.0f + __expf(-x)); }

// ------------- f32 -> f16 weight conversion with K padding to Kp ------------
__global__ void k_cvt_pad(const float* __restrict__ s, _Float16* __restrict__ d,
                          int K, int Kp, int cout) {
    int i = blockIdx.x * blockDim.x + threadIdx.x;      // over cout*Kp
    if (i >= cout * Kp) return;
    int co = i / Kp;
    int k  = i - co * Kp;
    d[i] = (k < K) ? (_Float16)s[(long)co * K + k] : (_Float16)0.0f;
}

// ---------------- input modulation: build x_new [B,T,9,HW] ------------------
__global__ void k_prep(const float* __restrict__ x, const float* __restrict__ ls,
                       const float* __restrict__ w1o3, const float* __restrict__ b1o3,
                       const float* __restrict__ w1u, const float* __restrict__ w2u,
                       const float* __restrict__ b1u, const float* __restrict__ b2u,
                       const float* __restrict__ w1v, const float* __restrict__ w2v,
                       const float* __restrict__ b1v, const float* __restrict__ b2v,
                       const float* __restrict__ w1t, const float* __restrict__ w2t,
                       const float* __restrict__ b1t, const float* __restrict__ b2t,
                       const float* __restrict__ w1f, const float* __restrict__ w2f,
                       const float* __restrict__ b1f, const float* __restrict__ b2f,
                       float* __restrict__ xnew)
{
    int i = blockIdx.x * blockDim.x + threadIdx.x;          // over B*T*HW
    if (i >= BB * TT * HW) return;
    int p  = i % HW;
    int bt = i / HW;
    int t  = bt % TT;
    int b  = bt / TT;
    float lr = ls[b * TT + t] * 0.017453292519943295f;
    const float* xb = x + ((long)(b * TT + t) * 5) * HW;
    float o3 = xb[0 * HW + p], u = xb[1 * HW + p], v = xb[2 * HW + p];
    float te = xb[3 * HW + p], fl = xb[4 * HW + p];
    float* o = xnew + ((long)(b * TT + t) * 9) * HW;
    o[0 * HW + p] = o3 + w1o3[p] * __sinf(lr + b1o3[p]);
    o[1 * HW + p] = u  * (w1u[p] * __sinf(lr + b1u[p]));
    o[2 * HW + p] = u  * (w2u[p] * __cosf(lr + b2u[p]));
    o[3 * HW + p] = v  * (w1v[p] * __sinf(lr + b1v[p]));
    o[4 * HW + p] = v  * (w2v[p] * __cosf(lr + b2v[p]));
    o[5 * HW + p] = te * (w1t[p] * __sinf(lr + b1t[p]));
    o[6 * HW + p] = te * (w2t[p] * __cosf(lr + b2t[p]));
    o[7 * HW + p] = fl * (w1f[p] * __sinf(lr + b1f[p]));
    o[8 * HW + p] = fl * (w2f[p] * __cosf(lr + b2f[p]));
}

// ---------------- WMMA implicit-GEMM convolution ----------------------------
// out[b, co, p] = bias[co] + sum_k  W[co, k] * im2col(src)[k, (b,p)]
// src may be a concat of two tensors (src0: cin0 ch, src1: cin1 ch).
// Block = 256 threads (8 wave32), computes 64(M) x 64(N) of output.
// Each wave: 16x32 sub-tile = 2x v_wmma_f32_16x16x32_f16 per K-step.
// Requirements: cout % 64 == 0, Kp % 32 == 0 (weights zero-padded to Kp).
template <int KSZ>
__global__ __launch_bounds__(256)
void k_conv_wmma(const float* __restrict__ src0, const float* __restrict__ src1,
                 int cin0, int cin1, long bs0, long bs1,
                 const _Float16* __restrict__ w, const float* __restrict__ bias,
                 float* __restrict__ out, int cout, int K, int Kp)
{
    __shared__ __align__(16) _Float16 Ash[64][40];   // [M rows][K=32 + pad]
    __shared__ __align__(16) _Float16 Bsh[64][40];   // [N cols][K=32 + pad]

    const int tid   = threadIdx.x;
    const int lane  = tid & 31;
    const int wid   = tid >> 5;
    const int mo    = (wid >> 1) * 16;     // wave M offset within tile (0/16/32/48)
    const int no    = (wid & 1) * 32;      // wave N offset within tile (0/32)
    const int tileN = blockIdx.x;
    const int tileM = blockIdx.y;

    constexpr int ks2 = KSZ * KSZ;
    constexpr int pad = KSZ >> 1;

    // B-tile gather coordinates (fixed per thread): col = tid&63, 8 k's each
    const int bcol = tid & 63;
    const int kgrp = tid >> 6;             // 0..3
    const int n    = tileN * 64 + bcol;
    const int nb   = n / HW;
    const int np   = n - nb * HW;
    const int py   = np / WW;
    const int px   = np - py * WW;
    const float* sb0 = src0 + (long)nb * bs0;
    const float* sb1 = src1 ? (src1 + (long)nb * bs1) : src0;

    // A-tile load coordinates: row = tid>>2, 8 contiguous k's (16B vector)
    const int arow = tid >> 2;
    const int akof = (tid & 3) * 8;
    const _Float16* wrow = w + (long)(tileM * 64 + arow) * Kp;

    v8f acc0 = {};
    v8f acc1 = {};

    for (int k0 = 0; k0 < Kp; k0 += 32) {
        __syncthreads();
        // ---- stage A: one unconditional 16B load per thread ----
        *(uint4*)&Ash[arow][akof] = *(const uint4*)&wrow[k0 + akof];
        if (k0 + 32 < Kp)
            __builtin_prefetch(&wrow[k0 + 32 + akof], 0, 1);

        // ---- stage B: im2col gather + f32->f16 convert ----
        #pragma unroll
        for (int j = 0; j < 8; ++j) {
            int kl = kgrp * 8 + j;
            int k  = k0 + kl;
            float bv = 0.0f;
            if (k < K) {
                int ci = k / ks2;                  // compile-time divisor
                int r  = k - ci * ks2;
                int ky = r / KSZ;
                int kx = r - ky * KSZ;
                int yy = py + ky - pad;
                int xx = px + kx - pad;
                if (yy >= 0 && yy < HH && xx >= 0 && xx < WW) {
                    const float* s = (ci < cin0) ? (sb0 + (long)ci * HW)
                                                 : (sb1 + (long)(ci - cin0) * HW);
                    bv = s[yy * WW + xx];
                }
            }
            Bsh[bcol][kl] = (_Float16)bv;
        }
        __syncthreads();

        // ---- load fragments (ISA 7.12.2 16-bit A layout, B symmetric) ----
        // lanes 0-15: K = 0..7 (v0-3) and 16..23 (v4-7); lanes 16-31: +8
        const int kb = (lane >> 4) * 8;
        const int lsub = lane & 15;
        v8h alo = *(const v8h*)&Ash[mo + lsub][kb];
        v8h ahi = *(const v8h*)&Ash[mo + lsub][kb + 16];
        v8h b0l = *(const v8h*)&Bsh[no + lsub][kb];
        v8h b0h = *(const v8h*)&Bsh[no + lsub][kb + 16];
        v8h b1l = *(const v8h*)&Bsh[no + 16 + lsub][kb];
        v8h b1h = *(const v8h*)&Bsh[no + 16 + lsub][kb + 16];
        v16h af  = __builtin_shufflevector(alo, ahi, 0,1,2,3,4,5,6,7,8,9,10,11,12,13,14,15);
        v16h bf0 = __builtin_shufflevector(b0l, b0h, 0,1,2,3,4,5,6,7,8,9,10,11,12,13,14,15);
        v16h bf1 = __builtin_shufflevector(b1l, b1h, 0,1,2,3,4,5,6,7,8,9,10,11,12,13,14,15);

        acc0 = __builtin_amdgcn_wmma_f32_16x16x32_f16(false, af, false, bf0,
                                                      (short)0, acc0, false, false);
        acc1 = __builtin_amdgcn_wmma_f32_16x16x32_f16(false, af, false, bf1,
                                                      (short)0, acc1, false, false);
    }

    // ---- epilogue: C/D layout (lanes 0-15: M=r, lanes 16-31: M=8+r) ----
    // cout is a multiple of 64 => no M guard; N covers exactly 144*64 => no N guard
    const int lr = lane & 15;
    const int lh = lane >> 4;
    const int n0 = tileN * 64 + no + lr;
    const int b0 = n0 / HW, p0 = n0 - b0 * HW;
    const int n1 = n0 + 16;
    const int b1 = n1 / HW, p1 = n1 - b1 * HW;
    #pragma unroll
    for (int r = 0; r < 8; ++r) {
        int co  = tileM * 64 + mo + lh * 8 + r;
        float bv = bias[co];
        out[((long)b0 * cout + co) * HW + p0] = acc0[r] + bv;
        out[((long)b1 * cout + co) * HW + p1] = acc1[r] + bv;
    }
}

// ---------------- LSTM gate elementwise -------------------------------------
// xc:[B,448,HW] hc:[B,256,HW] mc:[B,192,HW]; updates c,m in place, saves o_x+o_h
__global__ void k_gates(const float* __restrict__ xc, const float* __restrict__ hc,
                        const float* __restrict__ mc, float* __restrict__ c,
                        float* __restrict__ m, float* __restrict__ so)
{
    int i = blockIdx.x * blockDim.x + threadIdx.x;
    if (i >= BB * NHID * HW) return;
    int p  = i % HW;
    int ch = (i / HW) % NHID;
    int b  = i / (NHID * HW);
    const long xb = (long)b * 448 * HW, hb = (long)b * 256 * HW, mb = (long)b * 192 * HW;
    float i_x  = xc[xb + (0 * NHID + ch) * HW + p];
    float f_x  = xc[xb + (1 * NHID + ch) * HW + p];
    float g_x  = xc[xb + (2 * NHID + ch) * HW + p];
    float i_xp = xc[xb + (3 * NHID + ch) * HW + p];
    float f_xp = xc[xb + (4 * NHID + ch) * HW + p];
    float g_xp = xc[xb + (5 * NHID + ch) * HW + p];
    float o_x  = xc[xb + (6 * NHID + ch) * HW + p];
    float i_h  = hc[hb + (0 * NHID + ch) * HW + p];
    float f_h  = hc[hb + (1 * NHID + ch) * HW + p];
    float g_h  = hc[hb + (2 * NHID + ch) * HW + p];
    float o_h  = hc[hb + (3 * NHID + ch) * HW + p];
    float i_m  = mc[mb + (0 * NHID + ch) * HW + p];
    float f_m  = mc[mb + (1 * NHID + ch) * HW + p];
    float g_m  = mc[mb + (2 * NHID + ch) * HW + p];

    float it = sigf(i_x + i_h);
    float ft = sigf(f_x + f_h + 1.0f);
    float gt = tanhf(g_x + g_h);
    float cn = ft * c[i] + it * gt;
    float itp = sigf(i_xp + i_m);
    float ftp = sigf(f_xp + f_m + 1.0f);
    float gtp = tanhf(g_xp + g_m);
    float mn = ftp * m[i] + itp * gtp;
    c[i] = cn;
    m[i] = mn;
    so[i] = o_x + o_h;
}

// h_new = sigmoid(so + oc) * tanh(l)
__global__ void k_hout(const float* __restrict__ so, const float* __restrict__ oc,
                       const float* __restrict__ l, float* __restrict__ h)
{
    int i = blockIdx.x * blockDim.x + threadIdx.x;
    if (i >= BB * NHID * HW) return;
    h[i] = sigf(so[i] + oc[i]) * tanhf(l[i]);
}

// cur = x[:, TE-1, 0]
__global__ void k_init_cur(const float* __restrict__ x, float* __restrict__ cur)
{
    int i = blockIdx.x * blockDim.x + threadIdx.x;
    if (i >= BB * HW) return;
    int p = i % HW, b = i / HW;
    cur[i] = x[(((long)b * TT + (TE - 1)) * 5 + 0) * HW + p];
}

// decoder input: ch0 = cur + w1_o3*sin(ls+b1_o3); ch1..8 = x_new[:, t, 1:]
__global__ void k_dec_prep(const float* __restrict__ xnew, const float* __restrict__ ls,
                           const float* __restrict__ w1o3, const float* __restrict__ b1o3,
                           const float* __restrict__ cur, float* __restrict__ inp9, int t)
{
    int i = blockIdx.x * blockDim.x + threadIdx.x;
    if (i >= BB * HW) return;
    int p = i % HW, b = i / HW;
    float lr = ls[b * TT + t] * 0.017453292519943295f;
    inp9[((long)b * 9 + 0) * HW + p] = cur[i] + w1o3[p] * __sinf(lr + b1o3[p]);
    #pragma unroll
    for (int ch = 1; ch < 9; ++ch)
        inp9[((long)b * 9 + ch) * HW + p] = xnew[(((long)b * TT + t) * 9 + ch) * HW + p];
}

// pred = 1x1 conv (Cout=1) of h[2]; write to d_out slot + feedback buffer
__global__ void k_pred(const float* __restrict__ h2, const float* __restrict__ wl,
                       const float* __restrict__ bl, float* __restrict__ out,
                       float* __restrict__ cur, int step)
{
    int i = blockIdx.x * blockDim.x + threadIdx.x;
    if (i >= BB * HW) return;
    int p = i % HW, b = i / HW;
    float acc = bl[0];
    #pragma unroll
    for (int ch = 0; ch < NHID; ++ch)
        acc += h2[((long)b * NHID + ch) * HW + p] * wl[ch];
    out[((long)b * HORIZON + step) * HW + p] = acc;
    cur[i] = acc;
}

// ---------------------------------------------------------------------------
extern "C" void kernel_launch(void* const* d_in, const int* in_sizes, int n_in,
                              void* d_out, int out_size, void* d_ws, size_t ws_size,
                              hipStream_t stream)
{
    (void)in_sizes; (void)n_in; (void)out_size; (void)ws_size;
    const float* X    = (const float*)d_in[0];
    const float* LS   = (const float*)d_in[1];
    const float* W1O3 = (const float*)d_in[2];
    const float* B1O3 = (const float*)d_in[3];
    // modulation params: u(4..7), v(8..11), t(12..15), f(16..19) as w1,w2,b1,b2
    // cells: base 20 + 10*i : wx,bx,wh,bh,wm,bm,wo,bo,wl,bl
    const float* WLAST = (const float*)d_in[50];
    const float* BLAST = (const float*)d_in[51];

    // ---------------- workspace carve ----------------
    char* wsb = (char*)d_ws;
    auto carve = [&](size_t bytes) -> void* {
        void* pp = (void*)wsb;
        wsb += (bytes + 255) & ~(size_t)255;
        return pp;
    };
    const size_t SZS = (size_t)BB * NHID * HW;                       // one state tensor
    float* xnew = (float*)carve(sizeof(float) * (size_t)BB * TT * 9 * HW);
    float* st   = (float*)carve(sizeof(float) * 7 * SZS);            // h0..h2,c0..c2,m
    float* h[3] = { st, st + SZS, st + 2 * SZS };
    float* c[3] = { st + 3 * SZS, st + 4 * SZS, st + 5 * SZS };
    float* m    = st + 6 * SZS;
    float* xcb  = (float*)carve(sizeof(float) * (size_t)BB * 448 * HW);
    float* hcb  = (float*)carve(sizeof(float) * (size_t)BB * 256 * HW);
    float* mcb  = (float*)carve(sizeof(float) * (size_t)BB * 192 * HW);
    float* sob  = (float*)carve(sizeof(float) * SZS);
    float* ocb  = (float*)carve(sizeof(float) * SZS);
    float* lbb  = (float*)carve(sizeof(float) * SZS);
    float* inp9 = (float*)carve(sizeof(float) * (size_t)BB * 9 * HW);
    float* cur  = (float*)carve(sizeof(float) * (size_t)BB * HW);

    // K (logical) and Kp (padded to 32) per conv
    const int wxK [3] = { 81, 576, 576 };
    const int wxKp[3] = { 96, 576, 576 };
    const int whK = 576, whKp = 576;
    const int wmK = 576, wmKp = 576;
    const int woK = 1152, woKp = 1152;
    const int wlK = 128, wlKp = 128;

    _Float16* wf = (_Float16*)carve(sizeof(_Float16) * 1700000);
    _Float16 *wx16[3], *wh16[3], *wm16[3], *wo16[3], *wl16[3];
    {
        size_t off = 0;
        for (int i = 0; i < 3; ++i) {
            wx16[i] = wf + off; off += (size_t)448 * wxKp[i];
            wh16[i] = wf + off; off += (size_t)256 * whKp;
            wm16[i] = wf + off; off += (size_t)192 * wmKp;
            wo16[i] = wf + off; off += (size_t)64 * woKp;
            wl16[i] = wf + off; off += (size_t)64 * wlKp;
        }
    }

    // ---------------- weight conversion + input prep ----------------
    auto cvt = [&](const void* s, _Float16* dst, int K, int Kp, int cout) {
        int nel = cout * Kp;
        k_cvt_pad<<<(nel + 255) / 256, 256, 0, stream>>>((const float*)s, dst, K, Kp, cout);
    };
    for (int i = 0; i < 3; ++i) {
        int base = 20 + 10 * i;
        cvt(d_in[base + 0], wx16[i], wxK[i], wxKp[i], 448);
        cvt(d_in[base + 2], wh16[i], whK, whKp, 256);
        cvt(d_in[base + 4], wm16[i], wmK, wmKp, 192);
        cvt(d_in[base + 6], wo16[i], woK, woKp, 64);
        cvt(d_in[base + 8], wl16[i], wlK, wlKp, 64);
    }
    k_prep<<<(BB * TT * HW + 255) / 256, 256, 0, stream>>>(
        X, LS, W1O3, B1O3,
        (const float*)d_in[4],  (const float*)d_in[5],  (const float*)d_in[6],  (const float*)d_in[7],
        (const float*)d_in[8],  (const float*)d_in[9],  (const float*)d_in[10], (const float*)d_in[11],
        (const float*)d_in[12], (const float*)d_in[13], (const float*)d_in[14], (const float*)d_in[15],
        (const float*)d_in[16], (const float*)d_in[17], (const float*)d_in[18], (const float*)d_in[19],
        xnew);
    hipMemsetAsync(st, 0, sizeof(float) * 7 * SZS, stream);

    const int EW = (BB * NHID * HW + 255) / 256;   // elementwise grid over states
    const long HS = (long)NHID * HW;               // batch stride of state tensors

    auto run_cell = [&](int i, const float* inp, long inpStride, int cinInp) {
        int base = 20 + 10 * i;
        const float* bx = (const float*)d_in[base + 1];
        const float* bh = (const float*)d_in[base + 3];
        const float* bm = (const float*)d_in[base + 5];
        const float* bo = (const float*)d_in[base + 7];
        const float* bl = (const float*)d_in[base + 9];
        k_conv_wmma<3><<<dim3(NTILES, 7), 256, 0, stream>>>(
            inp, nullptr, cinInp, 0, inpStride, 0, wx16[i], bx, xcb, 448, wxK[i], wxKp[i]);
        k_conv_wmma<3><<<dim3(NTILES, 4), 256, 0, stream>>>(
            h[i], nullptr, NHID, 0, HS, 0, wh16[i], bh, hcb, 256, whK, whKp);
        k_conv_wmma<3><<<dim3(NTILES, 3), 256, 0, stream>>>(
            m, nullptr, NHID, 0, HS, 0, wm16[i], bm, mcb, 192, wmK, wmKp);
        k_gates<<<EW, 256, 0, stream>>>(xcb, hcb, mcb, c[i], m, sob);
        // mem = concat(c_new, m_new) handled via dual-source implicit GEMM
        k_conv_wmma<3><<<dim3(NTILES, 1), 256, 0, stream>>>(
            c[i], m, NHID, NHID, HS, HS, wo16[i], bo, ocb, 64, woK, woKp);
        k_conv_wmma<1><<<dim3(NTILES, 1), 256, 0, stream>>>(
            c[i], m, NHID, NHID, HS, HS, wl16[i], bl, lbb, 64, wlK, wlKp);
        k_hout<<<EW, 256, 0, stream>>>(sob, ocb, lbb, h[i]);
    };

    // ---------------- encoder ----------------
    for (int t = 0; t < TE; ++t) {
        const float* inp0 = xnew + (size_t)t * 9 * HW;   // [B,9,HW] slice, stride T*9*HW
        run_cell(0, inp0, (long)TT * 9 * HW, 9);
        run_cell(1, h[0], HS, NHID);
        run_cell(2, h[1], HS, NHID);
    }

    // ---------------- decoder ----------------
    const int PW = (BB * HW + 255) / 256;
    k_init_cur<<<PW, 256, 0, stream>>>(X, cur);
    for (int s = 0; s < HORIZON; ++s) {
        int t = TE + s;
        k_dec_prep<<<PW, 256, 0, stream>>>(xnew, LS, W1O3, B1O3, cur, inp9, t);
        run_cell(0, inp9, (long)9 * HW, 9);
        run_cell(1, h[0], HS, NHID);
        run_cell(2, h[1], HS, NHID);
        k_pred<<<PW, 256, 0, stream>>>(h[2], WLAST, BLAST, (float*)d_out, cur, s);
    }
}
